// HGGNet_33079838114116
// MI455X (gfx1250) — matrix operand order repaired and checked by reading, not compiled, not run on hardware
//
#include <hip/hip_runtime.h>
#include <cstddef>

typedef _Float16 f16_t;
typedef __attribute__((ext_vector_type(16))) _Float16 v16h;
typedef __attribute__((ext_vector_type(8)))  float    v8f;
typedef __attribute__((ext_vector_type(4)))  float    v4f;

#define KNN 16
#define BATCH 16

static inline int pad32(int x) { return (x + 31) & ~31; }

// ---------------------------------------------------------------------------
// Weight pre-pack: W[K,N] f32 -> f16 in WMMA B-fragment order, zero padded.
// Wp[((kt*tilesN + nt)*32 + lane)*16 + e] = W[kt*32 + (e/8)*16 + h*8 + e%8, nt*16 + r]
// ---------------------------------------------------------------------------
__global__ void pack_w_kernel(const float* __restrict__ W, f16_t* __restrict__ Wp,
                              int K, int N, int Np, size_t tot)
{
    size_t t = (size_t)blockIdx.x * blockDim.x + threadIdx.x;
    if (t >= tot) return;
    int e    = (int)(t & 15);
    int lane = (int)((t >> 4) & 31);
    size_t rest = t >> 9;
    int tilesN = Np >> 4;
    int nt = (int)(rest % tilesN);
    int kt = (int)(rest / tilesN);
    int r = lane & 15, h = lane >> 4;
    int k = (kt << 5) + ((e >> 3) << 4) + (h << 3) + (e & 7);
    int n = (nt << 4) + r;
    float v = (k < K && n < N) ? W[(size_t)k * N + n] : 0.0f;
    Wp[t] = (f16_t)v;
}

__global__ void pad_bias_kernel(const float* __restrict__ b, float* __restrict__ bp,
                                int N, int Np)
{
    int t = blockIdx.x * blockDim.x + threadIdx.x;
    if (t < Np) bp[t] = (t < N) ? b[t] : 0.0f;
}

// pts [M,3] -> padded [M,32] (zero fill) so GEMM A loads are unconditional
__global__ void pad3_kernel(const float* __restrict__ in, float* __restrict__ out, size_t tot)
{
    size_t t = (size_t)blockIdx.x * blockDim.x + threadIdx.x;
    if (t >= tot) return;
    int c = (int)(t & 31);
    size_t r = t >> 5;
    out[t] = (c < 3) ? in[r * 3 + c] : 0.0f;
}

// ---------------------------------------------------------------------------
// Branch-free WMMA GEMM with 4-way N-blocking per wave:
// out[M,Np] = act(A[M,Kp] * Wp + biasP); f16 multiply, f32 accumulate.
// M%16==0, Kp%32==0, Np%32==0 guaranteed by padding; pad lanes of Wp/biasP
// are zero so pad output columns are zero.  Each wave loads its A fragment
// once per k-step and feeds 4 WMMAs on consecutive packed-B fragments.
// All tile bookkeeping is forced into SGPRs via readfirstlane so every tile
// guard is a scalar branch: EXEC is never modified (WMMA requires EXEC=~0).
// ---------------------------------------------------------------------------
__global__ void gemm_fast_kernel(const float* __restrict__ A, const f16_t* __restrict__ Wp,
                                 const float* __restrict__ biasP, float* __restrict__ out,
                                 int M, int Kp, int Np, int relu)
{
    const int lane = threadIdx.x & 31;
    const int waveS = __builtin_amdgcn_readfirstlane(threadIdx.x >> 5);   // scalar wave id
    const int tilesN  = Np >> 4;
    const int tilesNG = (tilesN + 3) >> 2;                // groups of 4 N-tiles
    const int nwaves = blockDim.x >> 5;
    long long group = (long long)blockIdx.x * nwaves + waveS;             // scalar
    if (group >= (long long)(M >> 4) * tilesNG) return;   // scalar branch
    const int tm  = (int)(group / tilesNG);
    const int tg  = (int)(group % tilesNG);
    const int tn0 = tg << 2;
    const int ntc = (tilesN - tn0 < 4) ? (tilesN - tn0) : 4;              // scalar
    const int m0 = tm << 4;
    const int r = lane & 15, h = lane >> 4;

    const float* arow = A + (size_t)(m0 + r) * Kp;
    const f16_t* wb0 = Wp + ((size_t)tn0 * 32 + lane) * 16;
    const size_t wstep = (size_t)tilesN * 512;            // halfs per kt block

    v8f acc0 = {}, acc1 = {}, acc2 = {}, acc3 = {};
    const int nkt = Kp >> 5;
    for (int kt = 0; kt < nkt; ++kt) {
        const int k0 = kt << 5;
        const v4f* pa0 = (const v4f*)(arow + k0 + (h << 3));
        const v4f* pa1 = (const v4f*)(arow + k0 + 16 + (h << 3));
        v4f a0 = pa0[0], a1 = pa0[1];
        v4f a2 = pa1[0], a3 = pa1[1];
        v16h av;
        av[0]  = (f16_t)a0.x; av[1]  = (f16_t)a0.y; av[2]  = (f16_t)a0.z; av[3]  = (f16_t)a0.w;
        av[4]  = (f16_t)a1.x; av[5]  = (f16_t)a1.y; av[6]  = (f16_t)a1.z; av[7]  = (f16_t)a1.w;
        av[8]  = (f16_t)a2.x; av[9]  = (f16_t)a2.y; av[10] = (f16_t)a2.z; av[11] = (f16_t)a2.w;
        av[12] = (f16_t)a3.x; av[13] = (f16_t)a3.y; av[14] = (f16_t)a3.z; av[15] = (f16_t)a3.w;
        const f16_t* wk = wb0 + (size_t)kt * wstep;
        {
            v16h bv = *(const v16h*)(wk);
            acc0 = __builtin_amdgcn_wmma_f32_16x16x32_f16(false, av, false, bv, (short)0, acc0, false, false);
        }
        if (ntc > 1) {
            v16h bv = *(const v16h*)(wk + 512);
            acc1 = __builtin_amdgcn_wmma_f32_16x16x32_f16(false, av, false, bv, (short)0, acc1, false, false);
        }
        if (ntc > 2) {
            v16h bv = *(const v16h*)(wk + 1024);
            acc2 = __builtin_amdgcn_wmma_f32_16x16x32_f16(false, av, false, bv, (short)0, acc2, false, false);
        }
        if (ntc > 3) {
            v16h bv = *(const v16h*)(wk + 1536);
            acc3 = __builtin_amdgcn_wmma_f32_16x16x32_f16(false, av, false, bv, (short)0, acc3, false, false);
        }
    }
    #pragma unroll
    for (int j = 0; j < 4; ++j) {
        if (j >= ntc) break;                              // scalar branch
        v8f acc = (j == 0) ? acc0 : (j == 1) ? acc1 : (j == 2) ? acc2 : acc3;
        int n0 = (tn0 + j) << 4;
        float bcol = biasP[n0 + r];
        #pragma unroll
        for (int i = 0; i < 8; ++i) {
            float v = acc[i] + bcol;
            if (relu) v = fmaxf(v, 0.0f);
            out[(size_t)(m0 + i + (h << 3)) * Np + n0 + r] = v;
        }
    }
}

// ---------------------------------------------------------------------------
// kNN (K=16, self included) over 3D points. One thread per (b,i).
// ---------------------------------------------------------------------------
__global__ void knn16_kernel(const float* __restrict__ pts, int* __restrict__ idx,
                             int Bn, int N)
{
    size_t t = (size_t)blockIdx.x * blockDim.x + threadIdx.x;
    if (t >= (size_t)Bn * N) return;
    int b = (int)(t / N), i = (int)(t % N);
    const float* P = pts + (size_t)b * N * 3;
    float xi = P[i * 3], yi = P[i * 3 + 1], zi = P[i * 3 + 2];
    float bd[KNN]; int bi[KNN];
    for (int q = 0; q < KNN; ++q) { bd[q] = 3.0e38f; bi[q] = 0; }
    for (int m = 0; m < N; ++m) {
        float dx = P[m * 3] - xi, dy = P[m * 3 + 1] - yi, dz = P[m * 3 + 2] - zi;
        float d = dx * dx + dy * dy + dz * dz;
        if (d < bd[KNN - 1]) {
            int p = KNN - 1;
            while (p > 0 && bd[p - 1] > d) { bd[p] = bd[p - 1]; bi[p] = bi[p - 1]; --p; }
            bd[p] = d; bi[p] = m;
        }
    }
    for (int q = 0; q < KNN; ++q) idx[t * KNN + q] = bi[q];
}

// E[(b*N+n)*K+k, 0:Cp] = [center , neighbor - center, 0...] (padded stride Cp)
__global__ void build_edge_kernel(const float* __restrict__ feat, const int* __restrict__ idx,
                                  float* __restrict__ E, int N, int C, int fstride,
                                  int Cp, size_t tot)
{
    size_t t = (size_t)blockIdx.x * blockDim.x + threadIdx.x;
    if (t >= tot) return;
    int c = (int)(t % Cp);
    size_t row = t / Cp;         // (b*N+n)*K + k
    size_t pn  = row >> 4;       // b*N+n
    if (c >= 2 * C) { E[t] = 0.0f; return; }
    if (c < C) {
        E[t] = feat[pn * fstride + c];
    } else {
        int cc = c - C;
        int nb = idx[row];
        size_t b = pn / N;
        E[t] = feat[((size_t)b * N + nb) * fstride + cc] - feat[pn * fstride + cc];
    }
}

// out[pn, c] = max_k act[(pn*K + k)*sIn + c]   (in padded stride, out exact C)
__global__ void maxk_kernel(const float* __restrict__ act, float* __restrict__ out,
                            int C, int sIn, size_t tot)
{
    size_t t = (size_t)blockIdx.x * blockDim.x + threadIdx.x;
    if (t >= tot) return;
    int c = (int)(t % C);
    size_t pn = t / C;
    float m = -3.0e38f;
    for (int k = 0; k < KNN; ++k)
        m = fmaxf(m, act[(pn * KNN + k) * (size_t)sIn + c]);
    out[t] = m;
}

// g[b,c] = max_n f4[b,n,c]
__global__ void gmax_kernel(const float* __restrict__ f4, float* __restrict__ g,
                            int Npt, int C, size_t tot)
{
    size_t t = (size_t)blockIdx.x * blockDim.x + threadIdx.x;
    if (t >= tot) return;
    int c = (int)(t % C);
    size_t b = t / C;
    float m = -3.0e38f;
    for (int n = 0; n < Npt; ++n)
        m = fmaxf(m, f4[((size_t)b * Npt + n) * C + c]);
    g[t] = m;
}

// Farthest point sampling (one block per batch, serial iterations, LDS dist)
__global__ void fps_kernel(const float* __restrict__ pts, int* __restrict__ si,
                           int N, int m)
{
    const int b = blockIdx.x;
    const float* P = pts + (size_t)b * N * 3;
    int* S = si + (size_t)b * m;
    if (m >= N) {
        for (int i = threadIdx.x; i < m; i += blockDim.x) S[i] = (i < N) ? i : (N - 1);
        return;
    }
    __shared__ float dist[2048];
    __shared__ float rmx[256];
    __shared__ int   rai[256];
    __shared__ int   cur;
    for (int i = threadIdx.x; i < N; i += blockDim.x) dist[i] = 1.0e10f;
    if (threadIdx.x == 0) { cur = 0; S[0] = 0; }
    __syncthreads();
    for (int it = 1; it < m; ++it) {
        int ci = cur;
        float lx = P[ci * 3], ly = P[ci * 3 + 1], lz = P[ci * 3 + 2];
        float bm = -1.0f; int ba = 0;
        for (int i = threadIdx.x; i < N; i += blockDim.x) {
            float dx = P[i * 3] - lx, dy = P[i * 3 + 1] - ly, dz = P[i * 3 + 2] - lz;
            float d = dx * dx + dy * dy + dz * dz;
            float nd = fminf(dist[i], d);
            dist[i] = nd;
            if (nd > bm) { bm = nd; ba = i; }
        }
        rmx[threadIdx.x] = bm; rai[threadIdx.x] = ba;
        __syncthreads();
        for (int s = blockDim.x >> 1; s > 0; s >>= 1) {
            if (threadIdx.x < s) {
                float o = rmx[threadIdx.x + s];
                if (o > rmx[threadIdx.x] ||
                    (o == rmx[threadIdx.x] && rai[threadIdx.x + s] < rai[threadIdx.x])) {
                    rmx[threadIdx.x] = o; rai[threadIdx.x] = rai[threadIdx.x + s];
                }
            }
            __syncthreads();
        }
        if (threadIdx.x == 0) { cur = rai[0]; S[it] = rai[0]; }
        __syncthreads();
    }
}

// out[b,j,:] = in[b, si[b,j], :]
__global__ void gather_kernel(const float* __restrict__ in, const int* __restrict__ si,
                              float* __restrict__ out, int Nin, int m, int C, size_t tot)
{
    size_t t = (size_t)blockIdx.x * blockDim.x + threadIdx.x;
    if (t >= tot) return;
    int c = (int)(t % C);
    size_t j = (t / C) % m;
    size_t b = t / ((size_t)m * C);
    out[t] = in[((size_t)b * Nin + si[b * m + j]) * C + c];
}

__global__ void add3_bcast_kernel(const float* __restrict__ a, const float* __restrict__ p,
                                  const float* __restrict__ g, float* __restrict__ o,
                                  int Npt, int D, size_t tot)
{
    size_t t = (size_t)blockIdx.x * blockDim.x + threadIdx.x;
    if (t >= tot) return;
    int d = (int)(t % D);
    size_t b = t / ((size_t)Npt * D);
    o[t] = a[t] + p[t] + g[b * D + d];
}

__global__ void add2_kernel(const float* __restrict__ a, const float* __restrict__ b,
                            float* __restrict__ o, size_t n)
{
    size_t t = (size_t)blockIdx.x * blockDim.x + threadIdx.x;
    if (t < n) o[t] = a[t] + b[t];
}

// Attention: one block (256 thr) per (b,n).  Npt must equal blockDim.x.
__global__ void attn_kernel(const float* __restrict__ q, const float* __restrict__ k,
                            const float* __restrict__ v, float* __restrict__ out,
                            int Npt, int D)
{
    __shared__ float sc[256];
    __shared__ float red[256];
    int b = blockIdx.x / Npt;
    int n = blockIdx.x % Npt;
    int m = threadIdx.x;
    const float* qr = q + ((size_t)b * Npt + n) * D;
    const float* kr = k + ((size_t)b * Npt + m) * D;
    float s = 0.0f;
    for (int d = 0; d < D; ++d) s += qr[d] * kr[d];
    s *= rsqrtf((float)D);
    sc[m] = s; red[m] = s;
    __syncthreads();
    for (int st = 128; st > 0; st >>= 1) {
        if (m < st) red[m] = fmaxf(red[m], red[m + st]);
        __syncthreads();
    }
    float mx = red[0];
    __syncthreads();
    float e = __expf(sc[m] - mx);
    sc[m] = e; red[m] = e;
    __syncthreads();
    for (int st = 128; st > 0; st >>= 1) {
        if (m < st) red[m] += red[m + st];
        __syncthreads();
    }
    float inv = 1.0f / red[0];
    __syncthreads();
    sc[m] *= inv;
    __syncthreads();
    for (int d = threadIdx.x; d < D; d += blockDim.x) {
        float acc = 0.0f;
        for (int mm = 0; mm < Npt; ++mm)
            acc += sc[mm] * v[((size_t)b * Npt + mm) * D + d];
        out[((size_t)b * Npt + n) * D + d] = acc;
    }
}

// 3-NN inverse-distance interp; writes concat row [feat1 | interp(feat2)]
__global__ void interp_concat_kernel(const float* __restrict__ xyz1, const float* __restrict__ xyz2,
                                     const float* __restrict__ f1, const float* __restrict__ f2,
                                     float* __restrict__ E, int Na, int Nb, int C1, int C2)
{
    __shared__ int nn[3];
    __shared__ float ww[3];
    size_t b = blockIdx.x / Na;
    int i = blockIdx.x % Na;
    if (threadIdx.x == 0) {
        const float* p = xyz1 + (b * Na + i) * 3;
        float x = p[0], y = p[1], z = p[2];
        const float* qp = xyz2 + b * (size_t)Nb * 3;
        float d0 = 3e38f, d1 = 3e38f, d2 = 3e38f;
        int i0 = 0, i1 = 0, i2 = 0;
        for (int mI = 0; mI < Nb; ++mI) {
            float dx = qp[mI * 3] - x, dy = qp[mI * 3 + 1] - y, dz = qp[mI * 3 + 2] - z;
            float d = dx * dx + dy * dy + dz * dz;
            if (d < d0)      { d2 = d1; i2 = i1; d1 = d0; i1 = i0; d0 = d; i0 = mI; }
            else if (d < d1) { d2 = d1; i2 = i1; d1 = d;  i1 = mI; }
            else if (d < d2) { d2 = d;  i2 = mI; }
        }
        float w0 = 1.0f / fmaxf(d0, 1e-10f);
        float w1 = 1.0f / fmaxf(d1, 1e-10f);
        float w2 = 1.0f / fmaxf(d2, 1e-10f);
        float s = w0 + w1 + w2;
        nn[0] = i0; nn[1] = i1; nn[2] = i2;
        ww[0] = w0 / s; ww[1] = w1 / s; ww[2] = w2 / s;
    }
    __syncthreads();
    int Ct = C1 + C2;          // 768 / 384 -> already multiples of 32
    float* row = E + (size_t)blockIdx.x * Ct;
    const float* fr1 = f1 + (b * Na + i) * (size_t)C1;
    const float* base2 = f2 + b * (size_t)Nb * C2;
    for (int c = threadIdx.x; c < Ct; c += blockDim.x) {
        if (c < C1) row[c] = fr1[c];
        else {
            int cc = c - C1;
            row[c] = ww[0] * base2[(size_t)nn[0] * C2 + cc]
                   + ww[1] * base2[(size_t)nn[1] * C2 + cc]
                   + ww[2] * base2[(size_t)nn[2] * C2 + cc];
        }
    }
}

__global__ void concat2_kernel(const float* __restrict__ a, const float* __restrict__ b,
                               float* __restrict__ o, int Ca, int Cb, size_t tot)
{
    size_t t = (size_t)blockIdx.x * blockDim.x + threadIdx.x;
    if (t >= tot) return;
    int Ct = Ca + Cb;
    size_t r = t / Ct;
    int c = (int)(t % Ct);
    o[t] = (c < Ca) ? a[r * Ca + c] : b[r * Cb + (c - Ca)];
}

// rf row (padded to Cp) = [ g[b] (Dg) | head1[b,n] (Dh) | center[b,n] (3) | 0 ]
__global__ void build_rf_kernel(const float* __restrict__ g, const float* __restrict__ h,
                                const float* __restrict__ center, float* __restrict__ rf,
                                int N1pts, int Dg, int Dh, int Cp, size_t tot)
{
    size_t t = (size_t)blockIdx.x * blockDim.x + threadIdx.x;
    if (t >= tot) return;
    size_t row = t / Cp;
    int c = (int)(t % Cp);
    size_t b = row / N1pts;
    int n = (int)(row % N1pts);
    float v;
    if (c < Dg)                v = g[b * Dg + c];
    else if (c < Dg + Dh)      v = h[row * Dh + (c - Dg)];
    else if (c < Dg + Dh + 3)  v = center[((size_t)b * N1pts + n) * 3 + (c - Dg - Dh)];
    else                       v = 0.0f;
    rf[t] = v;
}

// X[m*16+s, 0:160] = [rf[m] (128) | grid(s) (2) | 0 pad]
__global__ void fold1_in_kernel(const float* __restrict__ rf, float* __restrict__ X, size_t tot)
{
    size_t t = (size_t)blockIdx.x * blockDim.x + threadIdx.x;
    if (t >= tot) return;
    int c = (int)(t % 160);
    size_t row = t / 160;
    size_t m = row >> 4;
    int s = (int)(row & 15);
    float v;
    if (c < 128) v = rf[m * 128 + c];
    else if (c < 130) {
        int gi = (c == 128) ? (s >> 2) : (s & 3);
        v = -1.0f + 2.0f * (float)gi / 3.0f;
    } else v = 0.0f;
    X[t] = v;
}

// X[row, 0:160] = [rf[m] (128) | y[row] (3, stride 32) | 0 pad]
__global__ void fold2_in_kernel(const float* __restrict__ rf, const float* __restrict__ y,
                                float* __restrict__ X, size_t tot)
{
    size_t t = (size_t)blockIdx.x * blockDim.x + threadIdx.x;
    if (t >= tot) return;
    int c = (int)(t % 160);
    size_t row = t / 160;
    size_t m = row >> 4;
    float v;
    if (c < 128)      v = rf[m * 128 + c];
    else if (c < 131) v = y[row * 32 + (c - 128)];
    else              v = 0.0f;
    X[t] = v;
}

// rebuild[m, s, c] = rel[(m*16+s)*32 + c] + center[m*3+c]
__global__ void rebuild_kernel(const float* __restrict__ rel, const float* __restrict__ center,
                               float* __restrict__ out, size_t tot)
{
    size_t t = (size_t)blockIdx.x * blockDim.x + threadIdx.x;
    if (t >= tot) return;
    size_t m = t / 48;
    int rem = (int)(t % 48);
    int s = rem / 3, c = rem % 3;
    out[t] = rel[(m * 16 + s) * 32 + c] + center[m * 3 + c];
}

// ---------------------------------------------------------------------------
// Host side
// ---------------------------------------------------------------------------
static inline unsigned gblk(size_t tot) { return (unsigned)((tot + 255) / 256); }

extern "C" void kernel_launch(void* const* d_in, const int* in_sizes, int n_in,
                              void* d_out, int out_size, void* d_ws, size_t ws_size,
                              hipStream_t stream)
{
    (void)in_sizes; (void)n_in; (void)out_size; (void)ws_size;

    // ---- unpack inputs (dict insertion-order flattening) ----
    int pi = 0;
    auto nx = [&]() -> const float* { return (const float*)d_in[pi++]; };
    const float* ptc = nx();
    const float* Wit = nx(); const float* Bit = nx();
    const float *Wds[2], *Bds[2]; for (int i = 0; i < 2; ++i) { Wds[i] = nx(); Bds[i] = nx(); }
    const float *We1[3], *Be1[3]; for (int i = 0; i < 3; ++i) { We1[i] = nx(); Be1[i] = nx(); }
    const float *We2[3], *Be2[3]; for (int i = 0; i < 3; ++i) { We2[i] = nx(); Be2[i] = nx(); }
    const float *We3[3], *Be3[3]; for (int i = 0; i < 3; ++i) { We3[i] = nx(); Be3[i] = nx(); }
    const float *We4[3], *Be4[3]; for (int i = 0; i < 3; ++i) { We4[i] = nx(); Be4[i] = nx(); }
    const float *Wco[2], *Bco[2]; for (int i = 0; i < 2; ++i) { Wco[i] = nx(); Bco[i] = nx(); }
    const float *Wp = nx(), *bp = nx(), *Wg = nx(), *bg = nx();
    const float *Wq = nx(), *bq = nx(), *Wk = nx(), *bk = nx();
    const float *Wv = nx(), *bv = nx(), *Wo = nx(), *bo = nx();
    const float *Wc1 = nx(), *Bc1 = nx();
    const float *Wc2 = nx(), *Bc2 = nx();
    const float *Wc3 = nx(), *Bc3 = nx();
    const float *Wc4 = nx(), *Bc4 = nx();
    const float *Wc5 = nx(), *Bc5 = nx();
    const float *Wfp32[2], *Bfp32[2]; for (int i = 0; i < 2; ++i) { Wfp32[i] = nx(); Bfp32[i] = nx(); }
    const float *Wfp21[2], *Bfp21[2]; for (int i = 0; i < 2; ++i) { Wfp21[i] = nx(); Bfp21[i] = nx(); }
    const float *Wfu1 = nx(), *Bfu1 = nx();
    const float *Wfu2 = nx(), *Bfu2 = nx();
    const float *Wred = nx(), *Bred = nx();
    const float *Wf1[3], *Bf1[3]; for (int i = 0; i < 3; ++i) { Wf1[i] = nx(); Bf1[i] = nx(); }
    const float *Wf2[3], *Bf2[3]; for (int i = 0; i < 3; ++i) { Wf2[i] = nx(); Bf2[i] = nx(); }

    // ---- workspace bump allocator (256B aligned blocks) ----
    char* wsp = (char*)d_ws;
    size_t off = 0;
    auto allocB = [&](size_t nbytes) -> void* { void* p = (void*)(wsp + off); off += (nbytes + 255) & ~(size_t)255; return p; };
    auto alloc  = [&](size_t nf) -> float* { return (float*)allocB(nf * 4); };
    auto alloci = [&](size_t ni) -> int*   { return (int*)allocB(ni * 4); };

    float* bufE = alloc(67108864);   // big staging (edge-E / concats / fold X)
    float* bufA = alloc(33554432);
    float* bufB = alloc(33554432);
    int* idx = alloci((size_t)BATCH * 2048 * KNN);
    int* si  = alloci((size_t)BATCH * 2048);
    float* ptcP = alloc(32768 * 32);
    float* f0   = alloc(32768 * 32);                      // stride 32 (Np of input_trans)
    float* o_ds = alloc(32768 * 8);
    float* f_ds = alloc(16384 * 8);   float* p_ds = alloc(16384 * 3);
    float* o_e1 = alloc(16384 * 128); float* f1 = alloc(16384 * 128); float* p1 = alloc(16384 * 3);
    float* o_e2 = alloc((size_t)16384 * 256); float* f2 = alloc(8192 * 256); float* p2 = alloc(8192 * 3);
    float* o_e3 = alloc((size_t)8192 * 512);  float* f3 = alloc(4096 * 512); float* p3 = alloc(4096 * 3);
    float* p3P = alloc(4096 * 32);
    float* f4 = alloc((size_t)4096 * 1024);
    float* gv = alloc(16 * 1024);
    float* Pp = alloc(4096 * 512); float* Gg = alloc(16 * 512);
    float* xb = alloc(4096 * 512); float* qb = alloc(4096 * 512);
    float* kb = alloc(4096 * 512); float* vb = alloc(4096 * 512);
    float* ao = alloc(4096 * 512); float* pj = alloc(4096 * 512);
    float* xo = alloc(4096 * 512); float* s3 = alloc(4096 * 512);
    float* s32 = alloc(8192 * 256); float* s2 = alloc(8192 * 256); float* s2f = alloc(8192 * 256);
    float* s21 = alloc(16384 * 128); float* s1 = alloc(16384 * 128); float* head1 = alloc(16384 * 128);
    float* gh = alloc(16 * 1024);
    float* rf = alloc(16384 * 128);
    float* f1out = alloc((size_t)262144 * 32);            // N=3 padded to 32
    float* rel   = alloc((size_t)262144 * 32);

    float* center = (float*)d_out;               // [16,1024,3] = [16,3072]
    float* rebuild = (float*)d_out + 49152;      // [16,16384,3]

    // ---- GEMM wrapper: pack weights to f16 fragments + padded bias, launch ----
    auto rungemm = [&](const float* A, const float* W, const float* bias, float* out,
                       int M, int K, int N, int relu) {
        int Kp = pad32(K), Np = pad32(N);
        f16_t* Wpk = (f16_t*)allocB((size_t)Kp * Np * 2);
        float* bpd = alloc((size_t)Np);
        size_t wt = (size_t)Kp * Np;
        pack_w_kernel<<<gblk(wt), 256, 0, stream>>>(W, Wpk, K, N, Np, wt);
        pad_bias_kernel<<<gblk((size_t)Np), 256, 0, stream>>>(bias, bpd, N, Np);
        int tilesNG = ((Np >> 4) + 3) >> 2;
        long long groups = (long long)(M / 16) * tilesNG;
        gemm_fast_kernel<<<(unsigned)((groups + 7) / 8), 256, 0, stream>>>(A, Wpk, bpd, out, M, Kp, Np, relu);
    };

    // ---- edge-conv driver ----
    auto edge_conv = [&](const float* feat, int fstride, const float* pts, int N, int Cin,
                         const float* const* Ws, const float* const* Bs,
                         const int* outs, int nL, float* outFeat) {
        int rows = BATCH * N * KNN;
        knn16_kernel<<<gblk((size_t)BATCH * N), 256, 0, stream>>>(pts, idx, BATCH, N);
        int Cp = pad32(2 * Cin);
        size_t tot = (size_t)rows * Cp;
        build_edge_kernel<<<gblk(tot), 256, 0, stream>>>(feat, idx, bufE, N, Cin, fstride, Cp, tot);
        float* tgt[3] = { bufA, bufB, bufE };
        const float* cur = bufE;
        int curC = 2 * Cin;
        for (int li = 0; li < nL; ++li) {
            float* dst = tgt[li];
            rungemm(cur, Ws[li], Bs[li], dst, rows, curC, outs[li], 1);
            cur = dst; curC = outs[li];
        }
        size_t mtot = (size_t)BATCH * N * curC;
        maxk_kernel<<<gblk(mtot), 256, 0, stream>>>(cur, outFeat, curC, pad32(curC), mtot);
    };
    auto downsample = [&](const float* feat, const float* pts, int N, int C, int m,
                          float* nf, float* np) {
        fps_kernel<<<BATCH, 256, 0, stream>>>(pts, si, N, m);
        size_t ft = (size_t)BATCH * m * C;
        gather_kernel<<<gblk(ft), 256, 0, stream>>>(feat, si, nf, N, m, C, ft);
        size_t pt = (size_t)BATCH * m * 3;
        gather_kernel<<<gblk(pt), 256, 0, stream>>>(pts, si, np, N, m, 3, pt);
    };

    // ---- backbone ----
    pad3_kernel<<<gblk((size_t)32768 * 32), 256, 0, stream>>>(ptc, ptcP, (size_t)32768 * 32);
    rungemm(ptcP, Wit, Bit, f0, BATCH * 2048, 3, 8, 0);                 // input_trans (f0 stride 32)
    int d_ds[2] = { 8, 8 };
    edge_conv(f0, 32, ptc, 2048, 8, Wds, Bds, d_ds, 2, o_ds);
    downsample(o_ds, ptc, 2048, 8, 1024, f_ds, p_ds);
    int d_e1[3] = { 64, 64, 128 };
    edge_conv(f_ds, 8, p_ds, 1024, 8, We1, Be1, d_e1, 3, o_e1);
    downsample(o_e1, p_ds, 1024, 128, 1024, f1, p1);                    // identity FPS
    int d_e2[3] = { 128, 128, 256 };
    edge_conv(f1, 128, p1, 1024, 128, We2, Be2, d_e2, 3, o_e2);
    downsample(o_e2, p1, 1024, 256, 512, f2, p2);
    int d_e3[3] = { 256, 256, 512 };
    edge_conv(f2, 256, p2, 512, 256, We3, Be3, d_e3, 3, o_e3);
    downsample(o_e3, p2, 512, 512, 256, f3, p3);
    int d_e4[3] = { 512, 512, 1024 };
    edge_conv(f3, 512, p3, 256, 512, We4, Be4, d_e4, 3, f4);
    gmax_kernel<<<gblk((size_t)BATCH * 1024), 256, 0, stream>>>(f4, gv, 256, 1024, (size_t)BATCH * 1024);

    // ---- dectrans + conv1 ----
    pad3_kernel<<<gblk((size_t)4096 * 32), 256, 0, stream>>>(p3, p3P, (size_t)4096 * 32);
    rungemm(p3P, Wp, bp, Pp, BATCH * 256, 3, 512, 0);
    rungemm(gv, Wg, bg, Gg, BATCH, 1024, 512, 0);
    size_t xt = (size_t)BATCH * 256 * 512;
    add3_bcast_kernel<<<gblk(xt), 256, 0, stream>>>(f3, Pp, Gg, xb, 256, 512, xt);
    rungemm(xb, Wq, bq, qb, BATCH * 256, 512, 512, 0);
    rungemm(xb, Wk, bk, kb, BATCH * 256, 512, 512, 0);
    rungemm(xb, Wv, bv, vb, BATCH * 256, 512, 512, 0);
    attn_kernel<<<BATCH * 256, 256, 0, stream>>>(qb, kb, vb, ao, 256, 512);
    rungemm(ao, Wo, bo, pj, BATCH * 256, 512, 512, 0);
    add2_kernel<<<gblk(xt), 256, 0, stream>>>(xb, pj, xo, xt);
    rungemm(xo, Wc1, Bc1, s3, BATCH * 256, 512, 512, 1);                // conv1

    // ---- fp_upsample 3->2, fuse1, conv3 ----
    interp_concat_kernel<<<BATCH * 512, 128, 0, stream>>>(p2, p3, f2, s3, bufE, 512, 256, 256, 512);
    rungemm(bufE, Wfp32[0], Bfp32[0], bufA, BATCH * 512, 768, 256, 1);
    rungemm(bufA, Wfp32[1], Bfp32[1], s32, BATCH * 512, 256, 256, 1);
    rungemm(f2, Wc2, Bc2, s2, BATCH * 512, 256, 256, 1);                // conv2
    size_t c1t = (size_t)BATCH * 512 * 512;
    concat2_kernel<<<gblk(c1t), 256, 0, stream>>>(s32, s2, bufE, 256, 256, c1t);
    rungemm(bufE, Wfu1, Bfu1, bufA, BATCH * 512, 512, 256, 1);          // fuse1
    rungemm(bufA, Wc3, Bc3, s2f, BATCH * 512, 256, 256, 1);             // conv3

    // ---- fp_upsample 2->1, fuse2, conv5 ----
    interp_concat_kernel<<<BATCH * 1024, 128, 0, stream>>>(p1, p2, f1, s2f, bufE, 1024, 512, 128, 256);
    rungemm(bufE, Wfp21[0], Bfp21[0], bufA, BATCH * 1024, 384, 128, 1);
    rungemm(bufA, Wfp21[1], Bfp21[1], s21, BATCH * 1024, 128, 128, 1);
    rungemm(f1, Wc4, Bc4, s1, BATCH * 1024, 128, 128, 1);               // conv4
    size_t c2t = (size_t)BATCH * 1024 * 256;
    concat2_kernel<<<gblk(c2t), 256, 0, stream>>>(s21, s1, bufE, 128, 128, c2t);
    rungemm(bufE, Wfu2, Bfu2, bufA, BATCH * 1024, 256, 128, 1);         // fuse2
    rungemm(bufA, Wc5, Bc5, head1, BATCH * 1024, 128, 128, 1);          // conv5

    // ---- coarse centers ----
    rungemm(gv, Wco[0], Bco[0], gh, BATCH, 1024, 1024, 1);
    rungemm(gh, Wco[1], Bco[1], center, BATCH, 1024, 3072, 0);          // -> d_out[0:49152]

    // ---- reduce + folding decoder ----
    int rfCp = pad32(1155);                                             // 1184
    size_t rft = (size_t)BATCH * 1024 * rfCp;
    build_rf_kernel<<<gblk(rft), 256, 0, stream>>>(gv, head1, center, bufE, 1024, 1024, 128, rfCp, rft);
    rungemm(bufE, Wred, Bred, rf, BATCH * 1024, 1155, 128, 0);

    const int MR = BATCH * 1024;           // 16384 folded rows
    size_t x1t = (size_t)MR * 16 * 160;
    fold1_in_kernel<<<gblk(x1t), 256, 0, stream>>>(rf, bufE, x1t);
    rungemm(bufE, Wf1[0], Bf1[0], bufA, MR * 16, 130, 128, 1);
    rungemm(bufA, Wf1[1], Bf1[1], bufB, MR * 16, 128, 64, 1);
    rungemm(bufB, Wf1[2], Bf1[2], f1out, MR * 16, 64, 3, 0);
    size_t x2t = (size_t)MR * 16 * 160;
    fold2_in_kernel<<<gblk(x2t), 256, 0, stream>>>(rf, f1out, bufE, x2t);
    rungemm(bufE, Wf2[0], Bf2[0], bufA, MR * 16, 131, 128, 1);
    rungemm(bufA, Wf2[1], Bf2[1], bufB, MR * 16, 128, 64, 1);
    rungemm(bufB, Wf2[2], Bf2[2], rel, MR * 16, 64, 3, 0);

    size_t rbt = (size_t)MR * 48;
    rebuild_kernel<<<gblk(rbt), 256, 0, stream>>>(rel, center, rebuild, rbt);
}
